// GlycanMS2Output_56049323213760
// MI455X (gfx1250) — matrix operand order repaired and checked by reading, not compiled, not run on hardware
//
#include <hip/hip_runtime.h>
#include <hip/hip_bf16.h>

#define F_DIM   256
#define T_DIM   8
#define N_MONO  150000
#define N_CLEAV 100000
#define N_FRAG  80000
#define N_OUT   40000

typedef __attribute__((ext_vector_type(8)))  _Float16 v8h;
typedef __attribute__((ext_vector_type(16))) _Float16 v16h;
typedef __attribute__((ext_vector_type(8)))  float    v8f;

__device__ __forceinline__ v16h cat8(v8h lo, v8h hi) {
  return __builtin_shufflevector(lo, hi, 0,1,2,3,4,5,6,7,8,9,10,11,12,13,14,15);
}

__device__ __forceinline__ v8f wmma_f16(v16h a, v16h b, v8f c) {
  // D = A(16x32 f16) x B(32x16 f16) + C(16x16 f32)
  return __builtin_amdgcn_wmma_f32_16x16x32_f16(false, a, false, b, (short)0, c,
                                                false, false);
}

__device__ __forceinline__ float sigm(float x) { return 1.f / (1.f + __expf(-x)); }

// ---------------------------------------------------------------------------
// Kernel 0: convert W_ih / W_hh to f16 (row-major [n][k], matches B layout),
// and precompute bsum = b_ih + b_hh.
// ---------------------------------------------------------------------------
__global__ __launch_bounds__(256) void prep_weights_kernel(
    const float* __restrict__ W_ih, const float* __restrict__ W_hh,
    const float* __restrict__ b_ih, const float* __restrict__ b_hh,
    _Float16* __restrict__ WihH, _Float16* __restrict__ WhhH,
    float* __restrict__ bsum)
{
  int idx = blockIdx.x * 256 + threadIdx.x;
  if (idx < 1024 * 512) WihH[idx] = (_Float16)W_ih[idx];
  if (idx < 1024 * 256) WhhH[idx] = (_Float16)W_hh[idx];
  if (idx < 1024)       bsum[idx] = b_ih[idx] + b_hh[idx];
}

// ---------------------------------------------------------------------------
// Kernel 1: attention-sum over 4-neighbor gathers (lost or retained side).
// One wave per cleavage node. Lane l owns feature columns [8l, 8l+8).
// Writes f16 result into cleav[node][colOff .. colOff+255].
// ---------------------------------------------------------------------------
__global__ __launch_bounds__(256) void attn_cleav_kernel(
    const float* __restrict__ feature, const float* __restrict__ Wg,
    const int* __restrict__ src, _Float16* __restrict__ cleav, int colOff)
{
  int node = blockIdx.x * 8 + (threadIdx.x >> 5);
  int lane = threadIdx.x & 31;
  if (node >= N_CLEAV) return;

  float wg[8];
  #pragma unroll
  for (int i = 0; i < 8; ++i) wg[i] = Wg[lane * 8 + i];

  float vals[4][8];
  float logit[4];
  #pragma unroll
  for (int d = 0; d < 4; ++d) {
    int s = src[node * 4 + d];
    const float* fp = feature + (size_t)s * F_DIM + lane * 8;
    float p = 0.f;
    #pragma unroll
    for (int i = 0; i < 8; ++i) { vals[d][i] = fp[i]; p += vals[d][i] * wg[i]; }
    #pragma unroll
    for (int off = 16; off > 0; off >>= 1) p += __shfl_xor(p, off, 32);
    logit[d] = p;
  }
  float mx = fmaxf(fmaxf(logit[0], logit[1]), fmaxf(logit[2], logit[3]));
  float e[4], sum = 0.f;
  #pragma unroll
  for (int d = 0; d < 4; ++d) { e[d] = __expf(logit[d] - mx); sum += e[d]; }
  float inv = 1.f / sum;

  v8h outv;
  #pragma unroll
  for (int i = 0; i < 8; ++i) {
    float o = 0.f;
    #pragma unroll
    for (int d = 0; d < 4; ++d) o += e[d] * inv * vals[d][i];
    outv[i] = (_Float16)o;
  }
  *(v8h*)(cleav + (size_t)node * 512 + colOff + lane * 8) = outv;
}

// ---------------------------------------------------------------------------
// Kernel 2: fused LSTM (2 steps) + attention over L=2 + output head.
// One block = 16 fragments. 8 waves, each wave owns 8 of the 64 N-tiles of the
// [16 x 1024] gate matrix, WMMA f16 K-loop over X (512) and H (256).
// ---------------------------------------------------------------------------
__global__ __launch_bounds__(256) void lstm_fused_kernel(
    const _Float16* __restrict__ cleav,   // [N_CLEAV][512]
    const int* __restrict__ join_src,     // [N_FRAG][2]
    const _Float16* __restrict__ WihH,    // [1024][512]
    const _Float16* __restrict__ WhhH,    // [1024][256]
    const float* __restrict__ bsum,       // [1024]
    const float* __restrict__ WgFrag,     // [256]
    const float* __restrict__ Wout,       // [256][8]
    const float* __restrict__ bout,       // [8]
    float* __restrict__ frag_out)         // [N_FRAG][8]
{
  __shared__ _Float16 sX[2][16][520];     // gathered inputs, padded rows
  __shared__ float    sG[16][1028];       // gate staging
  __shared__ _Float16 sHh1[16][264];      // h1 (f16: GEMM2 A + attention)
  __shared__ _Float16 sHh2[16][264];      // h2 (f16)
  __shared__ float    sC[16][260];        // c1, later reused as frag buffer
  __shared__ float    sLogit[16][2];

  const int tid   = threadIdx.x;
  const int lane  = tid & 31;
  const int w     = tid >> 5;
  const int mBase = blockIdx.x << 4;

  // A-operand addressing (16x32 f16 tile)
  const int arow   = lane & 15;
  const int asel   = (lane >> 4) << 4;    // K offset 0 or 16
  // B-operand addressing (32x16 f16 tile)
  const int bcol   = lane & 15;
  const int bko    = (lane >> 4) << 4;    // K offset 0 or 16
  // C/D addressing
  const int rowAdd = (lane >> 4) << 3;

  // ---- gather X0, X1 via join edges (2*16 rows x 512 f16) ----
  for (int i = tid; i < 2 * 16 * 64; i += 256) {
    int t = i >> 10;
    int r = (i >> 6) & 15;
    int c = i & 63;
    int srow = join_src[(mBase + r) * 2 + t];
    *(v8h*)&sX[t][r][c * 8] = *(const v8h*)(cleav + (size_t)srow * 512 + c * 8);
  }
  if (tid < 32) sLogit[tid >> 1][tid & 1] = 0.f;
  __syncthreads();

  // ---- GEMM1: gates = X0 @ W_ih^T ----
  for (int nt = w; nt < 64; nt += 8) {
    const int n0 = nt << 4;
    v8f acc = {};
    const _Float16* ap = &sX[0][arow][asel];
    const _Float16* bp = WihH + (size_t)(n0 + bcol) * 512 + bko;
    #pragma unroll
    for (int ks = 0; ks < 16; ++ks) {
      v8h lo = *(const v8h*)(ap + ks * 32);
      v8h hi = *(const v8h*)(ap + ks * 32 + 8);
      v16h b = *(const v16h*)(bp + ks * 32);
      acc = wmma_f16(cat8(lo, hi), b, acc);
    }
    #pragma unroll
    for (int r = 0; r < 8; ++r) sG[rowAdd + r][n0 + bcol] = acc[r];
  }
  __syncthreads();

  // ---- LSTM cell step 1 (c0 = h0 = 0 so f-gate term vanishes) ----
  for (int idx = tid; idx < 16 * 256; idx += 256) {
    int m = idx >> 8, j = idx & 255;
    float gi = sG[m][j]       + bsum[j];
    float gg = sG[m][512 + j] + bsum[512 + j];
    float go = sG[m][768 + j] + bsum[768 + j];
    float c  = sigm(gi) * tanhf(gg);
    float h  = sigm(go) * tanhf(c);
    sC[m][j]   = c;
    sHh1[m][j] = (_Float16)h;
  }
  __syncthreads();

  // ---- GEMM2: gates = X1 @ W_ih^T + H1 @ W_hh^T ----
  for (int nt = w; nt < 64; nt += 8) {
    const int n0 = nt << 4;
    v8f acc = {};
    const _Float16* ap = &sX[1][arow][asel];
    const _Float16* bp = WihH + (size_t)(n0 + bcol) * 512 + bko;
    #pragma unroll
    for (int ks = 0; ks < 16; ++ks) {
      v8h lo = *(const v8h*)(ap + ks * 32);
      v8h hi = *(const v8h*)(ap + ks * 32 + 8);
      v16h b = *(const v16h*)(bp + ks * 32);
      acc = wmma_f16(cat8(lo, hi), b, acc);
    }
    const _Float16* ah = &sHh1[arow][asel];
    const _Float16* bh = WhhH + (size_t)(n0 + bcol) * 256 + bko;
    #pragma unroll
    for (int ks = 0; ks < 8; ++ks) {
      v8h lo = *(const v8h*)(ah + ks * 32);
      v8h hi = *(const v8h*)(ah + ks * 32 + 8);
      v16h b = *(const v16h*)(bh + ks * 32);
      acc = wmma_f16(cat8(lo, hi), b, acc);
    }
    #pragma unroll
    for (int r = 0; r < 8; ++r) sG[rowAdd + r][n0 + bcol] = acc[r];
  }
  __syncthreads();

  // ---- LSTM cell step 2 ----
  for (int idx = tid; idx < 16 * 256; idx += 256) {
    int m = idx >> 8, j = idx & 255;
    float gi = sG[m][j]       + bsum[j];
    float gf = sG[m][256 + j] + bsum[256 + j];
    float gg = sG[m][512 + j] + bsum[512 + j];
    float go = sG[m][768 + j] + bsum[768 + j];
    float c  = sigm(gf) * sC[m][j] + sigm(gi) * tanhf(gg);
    float h  = sigm(go) * tanhf(c);
    sHh2[m][j] = (_Float16)h;
  }
  __syncthreads();

  // ---- attention logits over L=2: dot(h_l, Wg_frag) ----
  {
    int m = tid >> 4, jl = tid & 15;
    float p1 = 0.f, p2 = 0.f;
    #pragma unroll
    for (int k = 0; k < 16; ++k) {
      int j = jl + 16 * k;
      float wgv = WgFrag[j];
      p1 += (float)sHh1[m][j] * wgv;
      p2 += (float)sHh2[m][j] * wgv;
    }
    atomicAdd(&sLogit[m][0], p1);
    atomicAdd(&sLogit[m][1], p2);
  }
  __syncthreads();

  // ---- softmax(2) combine -> frag (reuse sC) ----
  for (int idx = tid; idx < 16 * 256; idx += 256) {
    int m = idx >> 8, j = idx & 255;
    float l1 = sLogit[m][0], l2 = sLogit[m][1];
    float mx = fmaxf(l1, l2);
    float e1 = __expf(l1 - mx), e2 = __expf(l2 - mx);
    float a1 = e1 / (e1 + e2);
    sC[m][j] = a1 * (float)sHh1[m][j] + (1.f - a1) * (float)sHh2[m][j];
  }
  __syncthreads();

  // ---- head: relu(frag @ W_out + b_out) -> [16][8] ----
  if (tid < 16 * 8) {
    int m = tid >> 3, u = tid & 7;
    float s = bout[u];
    for (int j = 0; j < 256; ++j) s += sC[m][j] * Wout[j * 8 + u];
    frag_out[(size_t)(mBase + m) * 8 + u] = fmaxf(s, 0.f);
  }
}

// ---------------------------------------------------------------------------
// Kernel 3: combine-edge gather-sum: out[n] = sum_d frag_out[combine_src[n,d]]
// ---------------------------------------------------------------------------
__global__ __launch_bounds__(256) void combine_kernel(
    const float* __restrict__ frag_out, const int* __restrict__ comb_src,
    float* __restrict__ out)
{
  int idx = blockIdx.x * 256 + threadIdx.x;
  if (idx >= N_OUT * T_DIM) return;
  int n = idx >> 3, u = idx & 7;
  float s = 0.f;
  #pragma unroll
  for (int d = 0; d < 4; ++d) {
    int f = comb_src[n * 4 + d];
    s += frag_out[(size_t)f * 8 + u];
  }
  out[idx] = s;
}

// ---------------------------------------------------------------------------
extern "C" void kernel_launch(void* const* d_in, const int* in_sizes, int n_in,
                              void* d_out, int out_size, void* d_ws, size_t ws_size,
                              hipStream_t stream) {
  const float* feature = (const float*)d_in[0];
  const float* Wg_lost = (const float*)d_in[1];
  const float* Wg_ret  = (const float*)d_in[2];
  const float* W_ih    = (const float*)d_in[3];
  const float* W_hh    = (const float*)d_in[4];
  const float* b_ih    = (const float*)d_in[5];
  const float* b_hh    = (const float*)d_in[6];
  const float* Wg_frag = (const float*)d_in[7];
  const float* W_out   = (const float*)d_in[8];
  const float* b_out   = (const float*)d_in[9];
  const int* lost_src  = (const int*)d_in[10];
  const int* ret_src   = (const int*)d_in[11];
  const int* join_src  = (const int*)d_in[12];
  const int* comb_src  = (const int*)d_in[13];

  char* ws = (char*)d_ws;
  size_t off = 0;
  _Float16* cleav = (_Float16*)(ws + off); off += (size_t)N_CLEAV * 512 * 2;  // 102.4 MB
  _Float16* WihH  = (_Float16*)(ws + off); off += 1024 * 512 * 2;
  _Float16* WhhH  = (_Float16*)(ws + off); off += 1024 * 256 * 2;
  float*    bsum  = (float*)(ws + off);    off += 1024 * 4;
  float*    fragv = (float*)(ws + off);    off += (size_t)N_FRAG * 8 * 4;

  prep_weights_kernel<<<2048, 256, 0, stream>>>(W_ih, W_hh, b_ih, b_hh,
                                                WihH, WhhH, bsum);
  attn_cleav_kernel<<<N_CLEAV / 8, 256, 0, stream>>>(feature, Wg_lost, lost_src,
                                                     cleav, 0);
  attn_cleav_kernel<<<N_CLEAV / 8, 256, 0, stream>>>(feature, Wg_ret, ret_src,
                                                     cleav, 256);
  lstm_fused_kernel<<<N_FRAG / 16, 256, 0, stream>>>(cleav, join_src, WihH, WhhH,
                                                     bsum, Wg_frag, W_out, b_out,
                                                     fragv);
  combine_kernel<<<(N_OUT * T_DIM + 255) / 256, 256, 0, stream>>>(fragv, comb_src,
                                                                  (float*)d_out);
}
